// LIFSNN_28089086116282
// MI455X (gfx1250) — matrix operand order repaired and checked by reading, not compiled, not run on hardware
//
#include <hip/hip_runtime.h>
#include <hip/hip_bf16.h>
#include <stdint.h>

// ---------------------------------------------------------------------------
// LIF SNN forward on gfx1250 (wave32, WMMA bf16 16x16x32, f32 accumulate)
// ---------------------------------------------------------------------------

typedef __bf16 bf16_t;
typedef __bf16 bf8v  __attribute__((ext_vector_type(8)));
typedef __bf16 bf16v __attribute__((ext_vector_type(16)));
typedef float  f8v   __attribute__((ext_vector_type(8)));

// ---- constants for this problem ----
#define BB   128          // batch
#define TT   512          // timesteps
#define JJ   700          // input features
#define JP   704          // padded input features (22 * 32)
#define HH   512          // hidden
#define OO   20           // readout channels
#define OP   32           // padded readout channels
#define NROWS (TT * BB)   // 65536 samples for BN / GEMM M-dim

// ---------------------------------------------------------------------------
// A-fragment loader: 16-bit A 16x32 layout (ISA 7.12.2):
//   lanes 0-15 : row M = lane,   K = {k0+0..7, k0+16..23}
//   lanes 16-31: row M = lane-16,K = {k0+8..15, k0+24..31}
// aRow already points at this lane's matrix row.
__device__ __forceinline__ bf16v load_a_frag(const bf16_t* __restrict__ aRow,
                                             int kb, int hi) {
    bf8v a0 = *(const bf8v*)(aRow + kb + hi * 8);
    bf8v a1 = *(const bf8v*)(aRow + kb + 16 + hi * 8);
    return __builtin_shufflevector(a0, a1, 0, 1, 2, 3, 4, 5, 6, 7,
                                            8, 9, 10, 11, 12, 13, 14, 15);
}

// B-fragment: column n of B == row n of original W[out,in] (row-major); a
// lane's 16 K-values match A:  W[col][k0 + hi*8 .. +7], W[col][k0 + 16 + hi*8 .. +7].
__device__ __forceinline__ bf16v load_b_frag(const bf16_t* __restrict__ bRow,
                                             int kb, int hi) {
    bf8v b0 = *(const bf8v*)(bRow + kb + hi * 8);
    bf8v b1 = *(const bf8v*)(bRow + kb + 16 + hi * 8);
    return __builtin_shufflevector(b0, b1, 0, 1, 2, 3, 4, 5, 6, 7,
                                            8, 9, 10, 11, 12, 13, 14, 15);
}

// ---------------------------------------------------------------------------
// GEMM1:  C[n, i] = sum_j X[b, t, jp] * W1[i, jp],  n = t*BB + b
//   X   : bf16 [BB, TT, JP]  (padded, zeros in jp >= JJ)
//   W   : bf16 [HH, JP]
//   C   : f32  [NROWS, HH]
// One wave -> 16x64 tile. 8 waves/block share tileM (A reuse in L1).
__global__ void __launch_bounds__(256)
lif_gemm1_kernel(const bf16_t* __restrict__ X,
                 const bf16_t* __restrict__ W,
                 float* __restrict__ C) {
    const int lane    = threadIdx.x & 31;
    const int wid     = threadIdx.x >> 5;
    const int laneLow = lane & 15;
    const int hi      = lane >> 4;

    const int tilesN = HH / 64;                 // 8
    const int waveId = blockIdx.x * 8 + wid;
    const int tileM  = waveId / tilesN;
    const int tileN  = waveId % tilesN;

    const int n0 = tileM * 16;
    const int t  = n0 / BB;                     // all 16 rows share t
    const int b0 = n0 % BB;

    const bf16_t* aRow = X + ((size_t)(b0 + laneLow) * TT + t) * JP;
    const bf16_t* bRow0 = W + (size_t)(tileN * 64 + 0 * 16 + laneLow) * JP;
    const bf16_t* bRow1 = W + (size_t)(tileN * 64 + 1 * 16 + laneLow) * JP;
    const bf16_t* bRow2 = W + (size_t)(tileN * 64 + 2 * 16 + laneLow) * JP;
    const bf16_t* bRow3 = W + (size_t)(tileN * 64 + 3 * 16 + laneLow) * JP;

    f8v acc0 = {}, acc1 = {}, acc2 = {}, acc3 = {};
    for (int kb = 0; kb < JP; kb += 32) {
        bf16v a = load_a_frag(aRow, kb, hi);
        bf16v b0f = load_b_frag(bRow0, kb, hi);
        bf16v b1f = load_b_frag(bRow1, kb, hi);
        bf16v b2f = load_b_frag(bRow2, kb, hi);
        bf16v b3f = load_b_frag(bRow3, kb, hi);
        acc0 = __builtin_amdgcn_wmma_f32_16x16x32_bf16(false, a, false, b0f, (short)0, acc0, false, false);
        acc1 = __builtin_amdgcn_wmma_f32_16x16x32_bf16(false, a, false, b1f, (short)0, acc1, false, false);
        acc2 = __builtin_amdgcn_wmma_f32_16x16x32_bf16(false, a, false, b2f, (short)0, acc2, false, false);
        acc3 = __builtin_amdgcn_wmma_f32_16x16x32_bf16(false, a, false, b3f, (short)0, acc3, false, false);
    }
    // C/D layout: VGPR v -> M = v (lanes 0-15) or 8+v (lanes 16-31), N = laneLow
    const int colBase = tileN * 64 + laneLow;
#pragma unroll
    for (int v = 0; v < 8; ++v) {
        const size_t row = (size_t)(n0 + v + hi * 8);
        C[row * HH + colBase + 0 * 16] = acc0[v];
        C[row * HH + colBase + 1 * 16] = acc1[v];
        C[row * HH + colBase + 2 * 16] = acc2[v];
        C[row * HH + colBase + 3 * 16] = acc3[v];
    }
}

// ---------------------------------------------------------------------------
// Generic GEMM: C[n, i] = sum_k A[n, k] * W[i, k]
//   A : bf16 [NROWS, K]   (row-major)
//   W : bf16 [>= NT*16*tilesN rows, K]
//   C : f32  [NROWS, ldc]
template <int NT>
__global__ void __launch_bounds__(256)
lif_gemm_nt_kernel(const bf16_t* __restrict__ A,
                   const bf16_t* __restrict__ W,
                   float* __restrict__ C,
                   int K, int ldc, int tilesN) {
    const int lane    = threadIdx.x & 31;
    const int wid     = threadIdx.x >> 5;
    const int laneLow = lane & 15;
    const int hi      = lane >> 4;

    const int waveId = blockIdx.x * 8 + wid;
    const int tileM  = waveId / tilesN;
    const int tileN  = waveId % tilesN;
    const int n0     = tileM * 16;

    const bf16_t* aRow = A + (size_t)(n0 + laneLow) * K;
    const bf16_t* bRow[NT];
#pragma unroll
    for (int j = 0; j < NT; ++j)
        bRow[j] = W + (size_t)(tileN * (NT * 16) + j * 16 + laneLow) * K;

    f8v acc[NT];
#pragma unroll
    for (int j = 0; j < NT; ++j) acc[j] = (f8v){};

    for (int kb = 0; kb < K; kb += 32) {
        bf16v a = load_a_frag(aRow, kb, hi);
#pragma unroll
        for (int j = 0; j < NT; ++j) {
            bf16v b = load_b_frag(bRow[j], kb, hi);
            acc[j] = __builtin_amdgcn_wmma_f32_16x16x32_bf16(false, a, false, b, (short)0, acc[j], false, false);
        }
    }
    const int colBase = tileN * (NT * 16) + laneLow;
#pragma unroll
    for (int j = 0; j < NT; ++j) {
#pragma unroll
        for (int v = 0; v < 8; ++v) {
            const size_t row = (size_t)(n0 + v + hi * 8);
            C[row * ldc + colBase + j * 16] = acc[j][v];
        }
    }
}

// ---------------------------------------------------------------------------
// x [BB,TT,JJ] f32 -> padded bf16 [BB,TT,JP] (zeros beyond JJ)
__global__ void lif_cvt_x_kernel(const float* __restrict__ X, bf16_t* __restrict__ Xp) {
    size_t idx = (size_t)blockIdx.x * blockDim.x + threadIdx.x;
    const size_t total = (size_t)BB * TT * JP;
    if (idx >= total) return;
    int jp = (int)(idx % JP);
    size_t bt = idx / JP;
    float v = (jp < JJ) ? X[bt * JJ + jp] : 0.0f;
    Xp[idx] = (bf16_t)v;
}

// W [CoIn, Kin] f32 -> padded bf16 [CoOut, Kout]
__global__ void lif_cvt_w_kernel(const float* __restrict__ Win, bf16_t* __restrict__ Wout,
                                 int CoIn, int Kin, int CoOut, int Kout) {
    size_t idx = (size_t)blockIdx.x * blockDim.x + threadIdx.x;
    size_t total = (size_t)CoOut * Kout;
    if (idx >= total) return;
    int k  = (int)(idx % Kout);
    int co = (int)(idx / Kout);
    float v = (co < CoIn && k < Kin) ? Win[(size_t)co * Kin + k] : 0.0f;
    Wout[idx] = (bf16_t)v;
}

// ---------------------------------------------------------------------------
// BN stats: per-channel sum and sum of squares over NROWS samples (atomics).
__global__ void lif_bn_stats_kernel(const float* __restrict__ H, int C,
                                    float* __restrict__ sum, float* __restrict__ sumsq) {
    int c = blockIdx.y * blockDim.x + threadIdx.x;
    if (c >= C) return;
    float s = 0.0f, q = 0.0f;
    for (int r = blockIdx.x; r < NROWS; r += gridDim.x) {
        float v = H[(size_t)r * C + c];
        s += v;
        q += v * v;
    }
    atomicAdd(&sum[c], s);
    atomicAdd(&sumsq[c], q);
}

// Fold BN into scale/shift; sigmoid(beta_raw).
__global__ void lif_bn_finalize_kernel(const float* __restrict__ sum,
                                       const float* __restrict__ sumsq,
                                       const float* __restrict__ gamma,
                                       const float* __restrict__ bias,
                                       const float* __restrict__ beta_raw,
                                       float* __restrict__ scale,
                                       float* __restrict__ shift,
                                       float* __restrict__ betaS,
                                       int C, int Cvalid) {
    int c = blockIdx.x * blockDim.x + threadIdx.x;
    if (c >= C) return;
    if (c >= Cvalid) { scale[c] = 0.f; shift[c] = 0.f; betaS[c] = 0.f; return; }
    const float invN = 1.0f / (float)NROWS;
    float m   = sum[c] * invN;
    float var = sumsq[c] * invN - m * m;
    float inv = rsqrtf(var + 1e-5f);
    float sc  = inv * gamma[c];
    scale[c]  = sc;
    shift[c]  = bias[c] - m * sc;
    betaS[c]  = 1.0f / (1.0f + expf(-beta_raw[c]));
}

// ---------------------------------------------------------------------------
// Hidden-layer LIF scan: one thread per (b, channel). Spikes -> bf16.
// U_t = beta*(U - S) + (1-beta)*xhat ; S = (U > 1)
__global__ void lif_scan_kernel(const float* __restrict__ H,    // [TT, BB*C]
                                const float* __restrict__ scale,
                                const float* __restrict__ shift,
                                const float* __restrict__ betaS,
                                bf16_t* __restrict__ Sout,
                                int BC, int C) {
    int e = blockIdx.x * blockDim.x + threadIdx.x;
    if (e >= BC) return;
    int c = e % C;
    float bt = betaS[c], sc = scale[c], sh = shift[c];
    float omb = 1.0f - bt;
    float U = 0.0f, S = 0.0f;
    for (int t = 0; t < TT; ++t) {
        size_t off = (size_t)t * BC + e;
        float xh = H[off] * sc + sh;
        U = bt * (U - S) + omb * xh;
        S = (U > 1.0f) ? 1.0f : 0.0f;
        Sout[off] = (bf16_t)S;
    }
}

// ---------------------------------------------------------------------------
// Readout: leaky integrate (no spike), softmax over 20 channels, sum over T.
// One thread per batch element; 20 membrane states live in registers.
__global__ void lif_readout_kernel(const float* __restrict__ HR,  // [TT*BB, OP]
                                   const float* __restrict__ scale,
                                   const float* __restrict__ shift,
                                   const float* __restrict__ betaS,
                                   float* __restrict__ out) {       // [BB, OO]
    int b = blockIdx.x * blockDim.x + threadIdx.x;
    if (b >= BB) return;
    float U[OO], accv[OO], sc[OO], sh[OO], bt[OO];
#pragma unroll
    for (int c = 0; c < OO; ++c) {
        U[c] = 0.0f; accv[c] = 0.0f;
        sc[c] = scale[c]; sh[c] = shift[c]; bt[c] = betaS[c];
    }
    for (int t = 0; t < TT; ++t) {
        const float* row = HR + ((size_t)t * BB + b) * OP;
        float m = -3.4e38f;
#pragma unroll
        for (int c = 0; c < OO; ++c) {
            float xh = row[c] * sc[c] + sh[c];
            U[c] = bt[c] * U[c] + (1.0f - bt[c]) * xh;
            m = fmaxf(m, U[c]);
        }
        float ssum = 0.0f;
        float ev[OO];
#pragma unroll
        for (int c = 0; c < OO; ++c) { ev[c] = __expf(U[c] - m); ssum += ev[c]; }
        float inv = 1.0f / ssum;
#pragma unroll
        for (int c = 0; c < OO; ++c) accv[c] += ev[c] * inv;
    }
#pragma unroll
    for (int c = 0; c < OO; ++c) out[(size_t)b * OO + c] = accv[c];
}

// ---------------------------------------------------------------------------
extern "C" void kernel_launch(void* const* d_in, const int* in_sizes, int n_in,
                              void* d_out, int out_size, void* d_ws, size_t ws_size,
                              hipStream_t stream) {
    const float* x     = (const float*)d_in[0];
    const float* W1    = (const float*)d_in[1];
    const float* beta1 = (const float*)d_in[2];
    const float* g1    = (const float*)d_in[3];
    const float* b1    = (const float*)d_in[4];
    const float* W2    = (const float*)d_in[5];
    const float* beta2 = (const float*)d_in[6];
    const float* g2    = (const float*)d_in[7];
    const float* b2    = (const float*)d_in[8];
    const float* Wr    = (const float*)d_in[9];
    const float* betar = (const float*)d_in[10];
    const float* gr    = (const float*)d_in[11];
    const float* br    = (const float*)d_in[12];
    float* out = (float*)d_out;

    // ---- workspace carve-up (aligned to 256B); total ~304 MB ----
    uint8_t* base = (uint8_t*)d_ws;
    size_t off = 0;
    auto carve = [&](size_t bytes) -> void* {
        void* p = base + off;
        off = (off + bytes + 255) & ~(size_t)255;
        return p;
    };
    bf16_t* xp  = (bf16_t*)carve((size_t)BB * TT * JP * 2);   // 92.3 MB
    bf16_t* w1b = (bf16_t*)carve((size_t)HH * JP * 2);        // 0.72 MB
    bf16_t* w2b = (bf16_t*)carve((size_t)HH * HH * 2);        // 0.5 MB
    bf16_t* wrb = (bf16_t*)carve((size_t)OP * HH * 2);        // 32 KB
    float*  h   = (float*)carve((size_t)NROWS * HH * 4);      // 134 MB
    bf16_t* s   = (bf16_t*)carve((size_t)NROWS * HH * 2);     // 67 MB
    float*  hr  = (float*)carve((size_t)NROWS * OP * 4);      // 8.4 MB
    float*  st1 = (float*)carve(2 * HH * 4);   // sum | sumsq
    float*  st2 = (float*)carve(2 * HH * 4);
    float*  st3 = (float*)carve(2 * OP * 4);
    float*  p1  = (float*)carve(3 * HH * 4);   // scale | shift | betaS
    float*  p2  = (float*)carve(3 * HH * 4);
    float*  p3  = (float*)carve(3 * OP * 4);

    // ---- precision conversion / padding ----
    {
        size_t tot = (size_t)BB * TT * JP;
        lif_cvt_x_kernel<<<(unsigned)((tot + 255) / 256), 256, 0, stream>>>(x, xp);
    }
    lif_cvt_w_kernel<<<(HH * JP + 255) / 256, 256, 0, stream>>>(W1, w1b, HH, JJ, HH, JP);
    lif_cvt_w_kernel<<<(HH * HH + 255) / 256, 256, 0, stream>>>(W2, w2b, HH, HH, HH, HH);
    lif_cvt_w_kernel<<<(OP * HH + 255) / 256, 256, 0, stream>>>(Wr, wrb, OO, HH, OP, HH);

    // ---- layer 1: GEMM -> BN stats -> fold -> LIF scan ----
    hipMemsetAsync(st1, 0, 2 * HH * 4, stream);
    lif_gemm1_kernel<<<NROWS / 16, 256, 0, stream>>>(xp, w1b, h);
    lif_bn_stats_kernel<<<dim3(256, HH / 256), 256, 0, stream>>>(h, HH, st1, st1 + HH);
    lif_bn_finalize_kernel<<<HH / 256, 256, 0, stream>>>(st1, st1 + HH, g1, b1, beta1,
                                                         p1, p1 + HH, p1 + 2 * HH, HH, HH);
    lif_scan_kernel<<<(BB * HH) / 256, 256, 0, stream>>>(h, p1, p1 + HH, p1 + 2 * HH, s, BB * HH, HH);

    // ---- layer 2 ----
    hipMemsetAsync(st2, 0, 2 * HH * 4, stream);
    lif_gemm_nt_kernel<4><<<(NROWS / 16) * (HH / 64) / 8, 256, 0, stream>>>(s, w2b, h, HH, HH, HH / 64);
    lif_bn_stats_kernel<<<dim3(256, HH / 256), 256, 0, stream>>>(h, HH, st2, st2 + HH);
    lif_bn_finalize_kernel<<<HH / 256, 256, 0, stream>>>(st2, st2 + HH, g2, b2, beta2,
                                                         p2, p2 + HH, p2 + 2 * HH, HH, HH);
    lif_scan_kernel<<<(BB * HH) / 256, 256, 0, stream>>>(h, p2, p2 + HH, p2 + 2 * HH, s, BB * HH, HH);

    // ---- readout layer: GEMM (padded to 32 cols) -> BN -> integrate+softmax ----
    hipMemsetAsync(st3, 0, 2 * OP * 4, stream);
    lif_gemm_nt_kernel<2><<<(NROWS / 16) / 8, 256, 0, stream>>>(s, wrb, hr, HH, OP, 1);
    lif_bn_stats_kernel<<<dim3(256, 1), 256, 0, stream>>>(hr, OP, st3, st3 + OP);
    lif_bn_finalize_kernel<<<1, OP, 0, stream>>>(st3, st3 + OP, gr, br, betar,
                                                 p3, p3 + OP, p3 + 2 * OP, OP, OO);
    lif_readout_kernel<<<1, BB, 0, stream>>>(hr, p3, p3 + OP, p3 + 2 * OP, out);

    (void)in_sizes; (void)n_in; (void)out_size; (void)ws_size;
}
